// RockfishEncoder_39333310497348
// MI455X (gfx1250) — matrix operand order repaired and checked by reading, not compiled, not run on hardware
//
#include <hip/hip_runtime.h>
#include <hip/hip_bf16.h>

// ---------------------------------------------------------------------------
// Rockfish encoder forward, CDNA5 (gfx1250) WMMA implementation.
// GEMMs: bf16 WMMA (v_wmma_f32_16x16x32_bf16), f32 accumulate.
// Global->LDS staging via Tensor Data Mover, double-buffered: tile k+1 DMA
// overlaps tile-k WMMA (s_wait_tensorcnt 2 -> in-order TDM completion).
// wave32, 256-thread blocks (8 waves).
// ---------------------------------------------------------------------------

#define B_   16
#define S_   512
#define T_   32
#define E_   512
#define A_   128
#define H_   8
#define HD_  64
#define FF_  2048
#define L_   4
#define EPS_ 1e-5f
#define SCALE_ 0.125f            // 1/sqrt(64)

typedef __attribute__((ext_vector_type(16))) __bf16 v16bf;
typedef __attribute__((ext_vector_type(8)))  float  v8f;
typedef __attribute__((ext_vector_type(4)))  unsigned int u32x4;
typedef __attribute__((ext_vector_type(8)))  int          i32x8;
typedef __attribute__((ext_vector_type(4)))  int          i32x4;

__device__ __forceinline__ unsigned short f2bf(float f) {
  union { float f; unsigned u; } c; c.f = f;
  unsigned r = (c.u + 0x7FFFu + ((c.u >> 16) & 1u)) >> 16;
  return (unsigned short)r;
}
__device__ __forceinline__ __bf16 bfbits(unsigned short u) {
  union { unsigned short u; __bf16 b; } c; c.u = u;
  return c.b;
}
__device__ __forceinline__ __bf16 f2b(float f) { return bfbits(f2bf(f)); }
__device__ __forceinline__ float gelu_exact(float x) {
  return 0.5f * x * (1.0f + erff(x * 0.70710678118654752f));
}
__device__ __forceinline__ unsigned lds_off(const void* p) {
  // LDS aperture: flat address low 32 bits are the LDS offset (ISA 10.2).
  return (unsigned)(unsigned long)p;
}

// ---------------------------------------------------------------------------
// TDM: async 2D fp32 tile load, global -> LDS.
//   tile_x (contiguous) x tile_y rows; row stride stride_x elements.
//   tensor_x/tensor_y: remaining tensor extent from tile origin (OOB -> 0).
//   LDS row padding: pad_interval_code (2^(c+1) DWORDs), pad_amount_code
//   (c+1 DWORDs appended).
// D# per CDNA5 ISA (count=1, type=2, data_size=4B)
// ---------------------------------------------------------------------------
__device__ __forceinline__ void tdm_load_2d(
    unsigned lds_addr, const void* gptr, int tile_x, int tile_y,
    long tensor_x, long tensor_y, long stride_x,
    int pad_interval_code, int pad_amount_code) {
  unsigned long ga = (unsigned long)gptr;
  u32x4 g0;
  g0[0] = 1u;                                        // count=1 (valid D#)
  g0[1] = lds_addr;                                  // lds_addr
  g0[2] = (unsigned)(ga & 0xFFFFFFFFu);              // global_addr[31:0]
  g0[3] = (unsigned)((ga >> 32) & 0x1FFFFFFu) | (2u << 30);  // [56:32] | type=2

  unsigned td0 = (unsigned)tensor_x;
  unsigned td1 = (unsigned)tensor_y;
  unsigned long s0 = (unsigned long)stride_x;
  i32x8 g1;
  g1[0] = (int)((2u << 16) | (1u << 20) |            // data_size=4B, pad_enable
                ((unsigned)pad_interval_code << 22) |
                ((unsigned)pad_amount_code << 25));
  g1[1] = (int)((td0 & 0xFFFFu) << 16);              // tensor_dim0[15:0]
  g1[2] = (int)((td0 >> 16) | ((td1 & 0xFFFFu) << 16));
  g1[3] = (int)((td1 >> 16) | ((unsigned)tile_x << 16));
  g1[4] = (int)((unsigned)tile_y & 0xFFFFu);         // tile_dim1, tile_dim2=0
  g1[5] = (int)(unsigned)(s0 & 0xFFFFFFFFu);         // dim0_stride[31:0]
  g1[6] = (int)(unsigned)((s0 >> 32) & 0xFFFFu);     // dim0_stride[47:32]
  g1[7] = 0;
  i32x4 z4 = {0, 0, 0, 0};
  i32x8 z8 = {0, 0, 0, 0, 0, 0, 0, 0};
  // 6-arg form (clang-23 / therock headers): groups 2/3 disabled (2D tensor)
  __builtin_amdgcn_tensor_load_to_lds(g0, g1, z4, z4, z8, 0);
}

// ---------------------------------------------------------------------------
// Universal batched GEMM:  C = epilogue( alpha * A @ B )  per batch z.
//   BNT=false: B is (K,N) row-major, ldb   (weights; TDM tile lands [k][n])
//   BNT=true : B is (N,K) row-major, ldb   (K-matrix of Q@K^T; tile [n][k])
// K multiple of 32 (true for all uses: 512, 2048, 128, 64, 32).
// Double-buffered TDM pipeline: DMA of tile k+1 overlaps compute of tile k.
// ---------------------------------------------------------------------------
template <bool BNT, bool BIAS, bool GELU, bool RESID>
__global__ __launch_bounds__(256) void gemm_ws(
    const float* __restrict__ Ag, const float* __restrict__ Bg,
    const float* __restrict__ biasg, const float* __restrict__ Rg,
    float* __restrict__ Cg,
    int M, int N, int K, int lda, int ldb, int ldc, int ldr,
    long sAo, long sAi, long sBo, long sBi, long sCo, long sCi,
    long sRo, long sRi, int innerN, float alpha) {
  __shared__ float As[2][64][34];  // A tile [m][k], 32-DW rows + 2-DW pad
  __shared__ float Bs[2][64][34];  // BNT: [n][k] rows+pad; else flat [k][66]

  const int z  = blockIdx.z;
  const int zo = z / innerN, zi = z - zo * innerN;
  const float* A = Ag + zo * sAo + zi * sAi;
  const float* B = Bg + zo * sBo + zi * sBi;
  float*       C = Cg + zo * sCo + zi * sCi;
  const float* R = RESID ? (Rg + zo * sRo + zi * sRi) : nullptr;

  const int tid  = threadIdx.x;
  const int lane = tid & 31;
  const int wave = tid >> 5;      // 0..7
  const int wm   = wave & 3;      // M sub-tile 0..3
  const int wn   = wave >> 2;     // N 32-block 0..1
  const int m0   = blockIdx.x * 64;
  const int n0   = blockIdx.y * 64;
  const int half = lane >> 4;
  const int l15  = lane & 15;

  // Issue both tile DMAs for K-step k0 into LDS buffer `buf` (wave 0 only).
  auto issue = [&](int k0, int buf) {
    tdm_load_2d(lds_off(&As[buf][0][0]), A + (long)m0 * lda + k0, 32, 64,
                (long)K - k0, (long)M - m0, lda, /*32 DW*/ 4, /*+2 DW*/ 1);
    if (BNT) {
      // B tile [n][k]: 64 rows x 32 cols from B[n0][k0]
      tdm_load_2d(lds_off(&Bs[buf][0][0]), B + (long)n0 * ldb + k0, 32, 64,
                  (long)K - k0, (long)N - n0, ldb, 4, 1);
    } else {
      // B tile [k][n]: 32 rows x 64 cols from B[k0][n0]; cols >= N -> 0.
      tdm_load_2d(lds_off(&Bs[buf][0][0]), B + (long)k0 * ldb + n0, 64, 32,
                  (long)N - n0, (long)K - k0, ldb, /*64 DW*/ 5, /*+2 DW*/ 1);
    }
  };

  v8f acc0 = {};
  v8f acc1 = {};

  if (tid < 32) issue(0, 0);       // preload tile 0 (outstanding = 2)
  int cur = 0;

  for (int k0 = 0; k0 < K; k0 += 32) {
    // ---- pipeline: issue next tile, then wait for current (in-order) ----
    if (tid < 32) {
      if (k0 + 32 < K) {
        issue(k0 + 32, cur ^ 1);                 // outstanding = 4
        __builtin_amdgcn_s_wait_tensorcnt(2);    // tile k done, k+1 in flight
      } else {
        __builtin_amdgcn_s_wait_tensorcnt(0);    // last tile: drain
      }
    }
    if (tid == 0 && k0 + 64 < K)
      __builtin_prefetch(A + (long)m0 * lda + k0 + 64, 0, 3);
    __syncthreads();

    // ---- fragments (contiguous k = 16*half + s per lane), fp32 LDS -> bf16 regs ----
    v16bf fa;
    {
      const int mr = wm * 16 + l15;
#pragma unroll
      for (int v2 = 0; v2 < 8; ++v2) {
        int kb = half * 16 + v2 * 2;
        fa[2 * v2]     = f2b(As[cur][mr][kb]);
        fa[2 * v2 + 1] = f2b(As[cur][mr][kb + 1]);
      }
    }
    v16bf fb0, fb1;
    {
      const int nr0 = wn * 32 + l15;
      const int nr1 = nr0 + 16;
      if (BNT) {
#pragma unroll
        for (int v2 = 0; v2 < 8; ++v2) {
          int kb = half * 16 + v2 * 2;      // B: contiguous K per lane-half
          fb0[2 * v2]     = f2b(Bs[cur][nr0][kb]);
          fb0[2 * v2 + 1] = f2b(Bs[cur][nr0][kb + 1]);
          fb1[2 * v2]     = f2b(Bs[cur][nr1][kb]);
          fb1[2 * v2 + 1] = f2b(Bs[cur][nr1][kb + 1]);
        }
      } else {
        const float* Bf = &Bs[cur][0][0];   // [k][66] rows
#pragma unroll
        for (int v2 = 0; v2 < 8; ++v2) {
          int kb = half * 16 + v2 * 2;
          fb0[2 * v2]     = f2b(Bf[kb * 66 + nr0]);
          fb0[2 * v2 + 1] = f2b(Bf[(kb + 1) * 66 + nr0]);
          fb1[2 * v2]     = f2b(Bf[kb * 66 + nr1]);
          fb1[2 * v2 + 1] = f2b(Bf[(kb + 1) * 66 + nr1]);
        }
      }
    }
    acc0 = __builtin_amdgcn_wmma_f32_16x16x32_bf16(false, fa, false, fb0,
                                                   (short)0, acc0, false, false);
    acc1 = __builtin_amdgcn_wmma_f32_16x16x32_bf16(false, fa, false, fb1,
                                                   (short)0, acc1, false, false);
    __syncthreads();   // next iteration re-issues into the buffer just read
    cur ^= 1;
  }

  // ---- epilogue: C/D layout = VGPR r: M = r + 8*half, N = lane&15 ----
#pragma unroll
  for (int r = 0; r < 8; ++r) {
    int mm  = m0 + wm * 16 + r + 8 * half;
    int nnA = n0 + wn * 32 + l15;
    int nnB = nnA + 16;
    if (mm < M) {
      float v0 = acc0[r] * alpha;
      float v1 = acc1[r] * alpha;
      if (BIAS)  { if (nnA < N) v0 += biasg[nnA]; if (nnB < N) v1 += biasg[nnB]; }
      if (GELU)  { v0 = gelu_exact(v0); v1 = gelu_exact(v1); }
      if (RESID) {
        if (nnA < N) v0 += R[(long)mm * ldr + nnA];
        if (nnB < N) v1 += R[(long)mm * ldr + nnB];
      }
      if (nnA < N) C[(long)mm * ldc + nnA] = v0;
      if (nnB < N) C[(long)mm * ldc + nnB] = v1;
    }
  }
}

// ---------------------------------------------------------------------------
// LayerNorm over last dim D (one row per block).
// ---------------------------------------------------------------------------
__global__ __launch_bounds__(256) void ln_rows(
    const float* __restrict__ x, const float* __restrict__ g,
    const float* __restrict__ b, float* __restrict__ y, int D) {
  long row = blockIdx.x;
  const float* xr = x + row * (long)D;
  float*       yr = y + row * (long)D;
  __shared__ float rA[8], rB[8];
  float s = 0.f, s2 = 0.f;
  for (int i = threadIdx.x; i < D; i += 256) { float v = xr[i]; s += v; s2 += v * v; }
#pragma unroll
  for (int o = 16; o; o >>= 1) { s += __shfl_xor(s, o, 32); s2 += __shfl_xor(s2, o, 32); }
  if ((threadIdx.x & 31) == 0) { rA[threadIdx.x >> 5] = s; rB[threadIdx.x >> 5] = s2; }
  __syncthreads();
  s = 0.f; s2 = 0.f;
#pragma unroll
  for (int i = 0; i < 8; ++i) { s += rA[i]; s2 += rB[i]; }
  float mean = s / D;
  float var  = s2 / D - mean * mean;
  float inv  = rsqrtf(var + EPS_);
  for (int i = threadIdx.x; i < D; i += 256)
    yr[i] = (xr[i] - mean) * inv * g[i] + b[i];
}

// ---------------------------------------------------------------------------
// InstanceNorm over (T,S) per (b,a) channel of (B,T,S,A).
// ---------------------------------------------------------------------------
__global__ __launch_bounds__(256) void inst_norm(
    const float* __restrict__ x, float* __restrict__ y) {
  int b = blockIdx.x / A_, a = blockIdx.x - b * A_;
  const long n = (long)T_ * S_;
  const float* xp = x + (long)b * n * A_ + a;
  float*       yp = y + (long)b * n * A_ + a;
  __shared__ float rA[8], rB[8];
  float s = 0.f, s2 = 0.f;
  for (long i = threadIdx.x; i < n; i += 256) { float v = xp[i * A_]; s += v; s2 += v * v; }
#pragma unroll
  for (int o = 16; o; o >>= 1) { s += __shfl_xor(s, o, 32); s2 += __shfl_xor(s2, o, 32); }
  if ((threadIdx.x & 31) == 0) { rA[threadIdx.x >> 5] = s; rB[threadIdx.x >> 5] = s2; }
  __syncthreads();
  s = 0.f; s2 = 0.f;
#pragma unroll
  for (int i = 0; i < 8; ++i) { s += rA[i]; s2 += rB[i]; }
  float mean = s / (float)n;
  float var  = s2 / (float)n - mean * mean;
  float inv  = rsqrtf(var + EPS_);
  for (long i = threadIdx.x; i < n; i += 256)
    yp[i * A_] = (xp[i * A_] - mean) * inv;
}

// ---------------------------------------------------------------------------
// Row softmax over scores (B,H,M,N); optional -1e9 key mask, optional
// alignment bias stored (B,M,N,H).
// ---------------------------------------------------------------------------
__global__ __launch_bounds__(256) void softmax_rows(
    float* __restrict__ sc, const int* __restrict__ mask,
    const float* __restrict__ alnbias, int Hn, int Mn, int Nn) {
  long row = blockIdx.x;                    // ((b*H + h)*M + m)
  int  m = (int)(row % Mn);
  long t = row / Mn;
  int  h = (int)(t % Hn);
  int  b = (int)(t / Hn);
  float* r = sc + row * (long)Nn;
  __shared__ float rA[8], rB[8];

  float mx = -3.4e38f;
  for (int i = threadIdx.x; i < Nn; i += 256) {
    float v = r[i];
    if (alnbias) v += alnbias[(((long)b * Mn + m) * Nn + i) * Hn + h];
    if (mask && mask[(long)b * Nn + i]) v += -1e9f;
    r[i] = v;
    mx = fmaxf(mx, v);
  }
#pragma unroll
  for (int o = 16; o; o >>= 1) mx = fmaxf(mx, __shfl_xor(mx, o, 32));
  if ((threadIdx.x & 31) == 0) rA[threadIdx.x >> 5] = mx;
  __syncthreads();
  mx = rA[0];
#pragma unroll
  for (int i = 1; i < 8; ++i) mx = fmaxf(mx, rA[i]);

  float sum = 0.f;
  for (int i = threadIdx.x; i < Nn; i += 256) {
    float e = __expf(r[i] - mx);
    r[i] = e;
    sum += e;
  }
#pragma unroll
  for (int o = 16; o; o >>= 1) sum += __shfl_xor(sum, o, 32);
  if ((threadIdx.x & 31) == 0) rB[threadIdx.x >> 5] = sum;
  __syncthreads();
  sum = 0.f;
#pragma unroll
  for (int i = 0; i < 8; ++i) sum += rB[i];
  float inv = 1.0f / sum;
  for (int i = threadIdx.x; i < Nn; i += 256) r[i] *= inv;
}

// ---------------------------------------------------------------------------
// Fused alignment block:
//   row r = (b,t,s):  a_row[k] = bM[b,t,k]*sS[b,s,k]  (0 if mask[b,s])
//   out[r,:] = alnIn[r,:] + gelu(a_row @ Wo + bo)      (Wo: A x A row-major)
// M = B*T*S (multiple of 64); a 64-row block spans one (b,t), s contiguous.
// ---------------------------------------------------------------------------
__global__ __launch_bounds__(256) void aln_gemm(
    const float* __restrict__ bM, const float* __restrict__ sS,
    const int* __restrict__ mask, const float* __restrict__ Wo,
    const float* __restrict__ bo, const float* __restrict__ alnIn,
    float* __restrict__ alnOut) {
  __shared__ unsigned short As[64][34];
  __shared__ unsigned short Bs[64][34];

  const int tid  = threadIdx.x;
  const int lane = tid & 31;
  const int wave = tid >> 5;
  const int wm   = wave & 3;
  const int wn   = wave >> 2;
  const int m0   = blockIdx.x * 64;
  const int n0   = blockIdx.y * 64;
  const int half = lane >> 4;
  const int l15  = lane & 15;

  // all 64 rows share (b,t); s = sBase..sBase+63  (T*S = 2^14, S = 2^9)
  const int bI = m0 >> 14;
  const int remI = m0 & 16383;
  const int tI = remI >> 9;
  const int sBase = remI & 511;
  const float* bRow = bM + ((long)bI * T_ + tI) * A_;
  const float* sRow = sS + ((long)bI * S_ + sBase) * A_;
  const int*   mRow = mask + (long)bI * S_ + sBase;

  v8f acc0 = {};
  v8f acc1 = {};

  for (int k0 = 0; k0 < A_; k0 += 32) {
    // A tile built on the fly from (b x s) outer product, branchless mask
#pragma unroll
    for (int t = 0; t < 8; ++t) {
      int idx = tid + t * 256;
      int r = idx >> 5, c = idx & 31;
      float v = bRow[k0 + c] * sRow[(long)r * A_ + (k0 + c)];
      v = mRow[r] ? 0.0f : v;
      As[r][c] = f2bf(v);
    }
    // Wo tile (K,N) row-major, staged as [n][k]
#pragma unroll
    for (int t = 0; t < 8; ++t) {
      int idx = tid + t * 256;
      int n = idx & 63, kk = idx >> 6;
      Bs[n][kk] = f2bf(Wo[(long)(k0 + kk) * A_ + (n0 + n)]);
    }
    __syncthreads();

    v16bf fa;
    {
      const int mr = wm * 16 + l15;
#pragma unroll
      for (int v2 = 0; v2 < 8; ++v2) {
        int kb = half * 16 + v2 * 2;
        fa[2 * v2]     = bfbits(As[mr][kb]);
        fa[2 * v2 + 1] = bfbits(As[mr][kb + 1]);
      }
    }
    v16bf fb0, fb1;
    {
      const int nr0 = wn * 32 + l15;
      const int nr1 = nr0 + 16;
#pragma unroll
      for (int v2 = 0; v2 < 8; ++v2) {
        int kb = half * 16 + v2 * 2;
        fb0[2 * v2]     = bfbits(Bs[nr0][kb]);
        fb0[2 * v2 + 1] = bfbits(Bs[nr0][kb + 1]);
        fb1[2 * v2]     = bfbits(Bs[nr1][kb]);
        fb1[2 * v2 + 1] = bfbits(Bs[nr1][kb + 1]);
      }
    }
    acc0 = __builtin_amdgcn_wmma_f32_16x16x32_bf16(false, fa, false, fb0,
                                                   (short)0, acc0, false, false);
    acc1 = __builtin_amdgcn_wmma_f32_16x16x32_bf16(false, fa, false, fb1,
                                                   (short)0, acc1, false, false);
    __syncthreads();
  }

#pragma unroll
  for (int r = 0; r < 8; ++r) {
    long mm = m0 + wm * 16 + r + 8 * half;
    int nnA = n0 + wn * 32 + l15;
    int nnB = nnA + 16;
    float v0 = gelu_exact(acc0[r] + bo[nnA]) + alnIn[mm * A_ + nnA];
    float v1 = gelu_exact(acc1[r] + bo[nnB]) + alnIn[mm * A_ + nnB];
    alnOut[mm * A_ + nnA] = v0;
    alnOut[mm * A_ + nnB] = v1;
  }
}

// ---------------------------------------------------------------------------
// Host-side orchestration
// ---------------------------------------------------------------------------
struct GA {
  const float *A, *B, *bias, *R;
  float* C;
  int M, N, K, lda, ldb, ldc, ldr;
  long sAo, sAi, sBo, sBi, sCo, sCi, sRo, sRi;
  int innerN, batch;
  float alpha;
};

template <bool BNT, bool BIAS, bool GELU, bool RESID>
static void run_gemm(const GA& a, hipStream_t st) {
  dim3 g((a.M + 63) / 64, (a.N + 63) / 64, a.batch);
  gemm_ws<BNT, BIAS, GELU, RESID><<<g, dim3(256), 0, st>>>(
      a.A, a.B, a.bias, a.R, a.C, a.M, a.N, a.K, a.lda, a.ldb, a.ldc, a.ldr,
      a.sAo, a.sAi, a.sBo, a.sBi, a.sCo, a.sCi, a.sRo, a.sRi, a.innerN,
      a.alpha);
}

static GA wgemm(const float* A, const float* W, const float* bias,
                const float* R, float* C, int M, int N, int K) {
  GA a{};
  a.A = A; a.B = W; a.bias = bias; a.R = R; a.C = C;
  a.M = M; a.N = N; a.K = K;
  a.lda = K; a.ldb = N; a.ldc = N; a.ldr = N;
  a.innerN = 1; a.batch = 1; a.alpha = 1.0f;
  return a;
}

extern "C" void kernel_launch(void* const* d_in, const int* in_sizes, int n_in,
                              void* d_out, int out_size, void* d_ws,
                              size_t ws_size, hipStream_t stream) {
  (void)in_sizes; (void)n_in; (void)out_size; (void)ws_size;

  const long BS = (long)B_ * S_, BT = (long)B_ * T_;
  const long BSE = BS * E_, BTE = BT * E_;
  const long BTS = (long)B_ * T_ * S_;
  const long BTSA = BTS * A_;
  const long SC = (long)B_ * H_ * S_ * S_;
  const long BSF = BS * FF_;

  // ---- workspace layout (floats) ----
  float* W = (float*)d_ws;
  long o = 0;
  auto alloc = [&](long n) { float* p = W + o; o += n; return p; };
  float* alnA  = alloc(BTSA);
  float* alnB  = alloc(BTSA);
  float* instb = alloc(BTSA);
  float* scores = alloc(SC);
  float* y1  = alloc(BSE);
  float* qb  = alloc(BSE);
  float* kb  = alloc(BSE);
  float* vb  = alloc(BSE);
  float* ob  = alloc(BSE);
  float* hb  = alloc(BSF);
  float* nbb = alloc(BTE);
  float* nsb = alloc(BSE);
  float* sbuf = alloc(BS * A_);
  float* bMb  = alloc(BT * A_);
  float* abias = alloc(BTS * H_);

  // ---- running state in d_out: [signal | bases | alignment] ----
  float* sig = (float*)d_out;
  float* bas = sig + BSE;
  float* alnFinal = bas + BTE;
  (void)hipMemcpyAsync(sig, d_in[0], (size_t)BSE * 4, hipMemcpyDeviceToDevice, stream);
  (void)hipMemcpyAsync(bas, d_in[1], (size_t)BTE * 4, hipMemcpyDeviceToDevice, stream);

  const int* maskp = (const int*)d_in[63];
  const float* alnCur = (const float*)d_in[2];

  const long EE = (long)E_ * E_;

  for (int i = 0; i < L_; ++i) {
    auto P = [&](int idx, long per) -> const float* {
      return (const float*)d_in[idx] + (long)i * per;
    };
    float* alnNext = (i == L_ - 1) ? alnFinal : ((i % 2 == 0) ? alnA : alnB);

    // nb = ln(bases, bases_norm)
    ln_rows<<<dim3((unsigned)BT), dim3(256), 0, stream>>>(bas, P(51, E_), P(52, E_), nbb, E_);

    // =================== SIGNAL decoder layer ('sig') ===================
    // --- self attention ---
    ln_rows<<<dim3((unsigned)BS), dim3(256), 0, stream>>>(sig, P(3, E_), P(4, E_), y1, E_);
    run_gemm<false, true, false, false>(wgemm(y1, P(5, EE), P(9, E_), nullptr, qb, (int)BS, E_, E_), stream);
    run_gemm<false, true, false, false>(wgemm(y1, P(6, EE), P(10, E_), nullptr, kb, (int)BS, E_, E_), stream);
    run_gemm<false, true, false, false>(wgemm(y1, P(7, EE), P(11, E_), nullptr, vb, (int)BS, E_, E_), stream);
    {
      GA a{}; a.A = qb; a.B = kb; a.C = scores;
      a.M = S_; a.N = S_; a.K = HD_; a.lda = E_; a.ldb = E_; a.ldc = S_;
      a.sAo = (long)S_ * E_; a.sAi = HD_; a.sBo = (long)S_ * E_; a.sBi = HD_;
      a.sCo = (long)H_ * S_ * S_; a.sCi = (long)S_ * S_;
      a.innerN = H_; a.batch = B_ * H_; a.alpha = SCALE_;
      run_gemm<true, false, false, false>(a, stream);
    }
    softmax_rows<<<dim3((unsigned)(B_ * H_ * S_)), dim3(256), 0, stream>>>(scores, maskp, nullptr, H_, S_, S_);
    {
      GA a{}; a.A = scores; a.B = vb; a.C = ob;
      a.M = S_; a.N = HD_; a.K = S_; a.lda = S_; a.ldb = E_; a.ldc = E_;
      a.sAo = (long)H_ * S_ * S_; a.sAi = (long)S_ * S_;
      a.sBo = (long)S_ * E_; a.sBi = HD_;
      a.sCo = (long)S_ * E_; a.sCi = HD_;
      a.innerN = H_; a.batch = B_ * H_; a.alpha = 1.0f;
      run_gemm<false, false, false, false>(a, stream);
    }
    run_gemm<false, true, false, true>(wgemm(ob, P(8, EE), P(12, E_), sig, sig, (int)BS, E_, E_), stream);

    // --- cross attention vs nb (no bias terms) ---
    ln_rows<<<dim3((unsigned)BS), dim3(256), 0, stream>>>(sig, P(13, E_), P(14, E_), y1, E_);
    run_gemm<false, false, false, false>(wgemm(y1, P(15, EE), nullptr, nullptr, qb, (int)BS, E_, E_), stream);
    run_gemm<false, false, false, false>(wgemm(nbb, P(16, EE), nullptr, nullptr, kb, (int)BT, E_, E_), stream);
    run_gemm<false, false, false, false>(wgemm(nbb, P(17, EE), nullptr, nullptr, vb, (int)BT, E_, E_), stream);
    {
      GA a{}; a.A = qb; a.B = kb; a.C = scores;
      a.M = S_; a.N = T_; a.K = HD_; a.lda = E_; a.ldb = E_; a.ldc = T_;
      a.sAo = (long)S_ * E_; a.sAi = HD_; a.sBo = (long)T_ * E_; a.sBi = HD_;
      a.sCo = (long)H_ * S_ * T_; a.sCi = (long)S_ * T_;
      a.innerN = H_; a.batch = B_ * H_; a.alpha = SCALE_;
      run_gemm<true, false, false, false>(a, stream);
    }
    softmax_rows<<<dim3((unsigned)(B_ * H_ * S_)), dim3(256), 0, stream>>>(scores, nullptr, nullptr, H_, S_, T_);
    {
      GA a{}; a.A = scores; a.B = vb; a.C = ob;
      a.M = S_; a.N = HD_; a.K = T_; a.lda = T_; a.ldb = E_; a.ldc = E_;
      a.sAo = (long)H_ * S_ * T_; a.sAi = (long)S_ * T_;
      a.sBo = (long)T_ * E_; a.sBi = HD_;
      a.sCo = (long)S_ * E_; a.sCi = HD_;
      a.innerN = H_; a.batch = B_ * H_; a.alpha = 1.0f;
      run_gemm<false, false, false, false>(a, stream);
    }
    run_gemm<false, true, false, true>(wgemm(ob, P(18, EE), P(20, E_), sig, sig, (int)BS, E_, E_), stream);

    // --- feed forward ---
    ln_rows<<<dim3((unsigned)BS), dim3(256), 0, stream>>>(sig, P(21, E_), P(22, E_), y1, E_);
    run_gemm<false, true, true, false>(wgemm(y1, P(23, (long)E_ * FF_), P(24, FF_), nullptr, hb, (int)BS, FF_, E_), stream);
    run_gemm<false, true, false, true>(wgemm(hb, P(25, (long)FF_ * E_), P(26, E_), sig, sig, (int)BS, E_, FF_), stream);

    // ns = ln(signal, signal_norm)
    ln_rows<<<dim3((unsigned)BS), dim3(256), 0, stream>>>(sig, P(53, E_), P(54, E_), nsb, E_);

    // =================== BASES decoder layer ('base') ===================
    // --- self attention (no mask) ---
    ln_rows<<<dim3((unsigned)BT), dim3(256), 0, stream>>>(bas, P(27, E_), P(28, E_), y1, E_);
    run_gemm<false, true, false, false>(wgemm(y1, P(29, EE), P(33, E_), nullptr, qb, (int)BT, E_, E_), stream);
    run_gemm<false, true, false, false>(wgemm(y1, P(30, EE), P(34, E_), nullptr, kb, (int)BT, E_, E_), stream);
    run_gemm<false, true, false, false>(wgemm(y1, P(31, EE), P(35, E_), nullptr, vb, (int)BT, E_, E_), stream);
    {
      GA a{}; a.A = qb; a.B = kb; a.C = scores;
      a.M = T_; a.N = T_; a.K = HD_; a.lda = E_; a.ldb = E_; a.ldc = T_;
      a.sAo = (long)T_ * E_; a.sAi = HD_; a.sBo = (long)T_ * E_; a.sBi = HD_;
      a.sCo = (long)H_ * T_ * T_; a.sCi = (long)T_ * T_;
      a.innerN = H_; a.batch = B_ * H_; a.alpha = SCALE_;
      run_gemm<true, false, false, false>(a, stream);
    }
    softmax_rows<<<dim3((unsigned)(B_ * H_ * T_)), dim3(256), 0, stream>>>(scores, nullptr, nullptr, H_, T_, T_);
    {
      GA a{}; a.A = scores; a.B = vb; a.C = ob;
      a.M = T_; a.N = HD_; a.K = T_; a.lda = T_; a.ldb = E_; a.ldc = E_;
      a.sAo = (long)H_ * T_ * T_; a.sAi = (long)T_ * T_;
      a.sBo = (long)T_ * E_; a.sBi = HD_;
      a.sCo = (long)T_ * E_; a.sCi = HD_;
      a.innerN = H_; a.batch = B_ * H_; a.alpha = 1.0f;
      run_gemm<false, false, false, false>(a, stream);
    }
    run_gemm<false, true, false, true>(wgemm(ob, P(32, EE), P(36, E_), bas, bas, (int)BT, E_, E_), stream);

    // --- cross attention vs ns, with instnorm(aln)@Waln bias + mask ---
    inst_norm<<<dim3((unsigned)(B_ * A_)), dim3(256), 0, stream>>>(alnCur, instb);
    run_gemm<false, false, false, false>(wgemm(instb, P(43, (long)A_ * H_), nullptr, nullptr, abias, (int)BTS, H_, A_), stream);
    ln_rows<<<dim3((unsigned)BT), dim3(256), 0, stream>>>(bas, P(37, E_), P(38, E_), y1, E_);
    run_gemm<false, false, false, false>(wgemm(y1, P(39, EE), nullptr, nullptr, qb, (int)BT, E_, E_), stream);
    run_gemm<false, false, false, false>(wgemm(nsb, P(40, EE), nullptr, nullptr, kb, (int)BS, E_, E_), stream);
    run_gemm<false, false, false, false>(wgemm(nsb, P(41, EE), nullptr, nullptr, vb, (int)BS, E_, E_), stream);
    {
      GA a{}; a.A = qb; a.B = kb; a.C = scores;
      a.M = T_; a.N = S_; a.K = HD_; a.lda = E_; a.ldb = E_; a.ldc = S_;
      a.sAo = (long)T_ * E_; a.sAi = HD_; a.sBo = (long)S_ * E_; a.sBi = HD_;
      a.sCo = (long)H_ * T_ * S_; a.sCi = (long)T_ * S_;
      a.innerN = H_; a.batch = B_ * H_; a.alpha = SCALE_;
      run_gemm<true, false, false, false>(a, stream);
    }
    softmax_rows<<<dim3((unsigned)(B_ * H_ * T_)), dim3(256), 0, stream>>>(scores, maskp, abias, H_, T_, S_);
    {
      GA a{}; a.A = scores; a.B = vb; a.C = ob;
      a.M = T_; a.N = HD_; a.K = S_; a.lda = S_; a.ldb = E_; a.ldc = E_;
      a.sAo = (long)H_ * T_ * S_; a.sAi = (long)T_ * S_;
      a.sBo = (long)S_ * E_; a.sBi = HD_;
      a.sCo = (long)T_ * E_; a.sCi = HD_;
      a.innerN = H_; a.batch = B_ * H_; a.alpha = 1.0f;
      run_gemm<false, false, false, false>(a, stream);
    }
    run_gemm<false, true, false, true>(wgemm(ob, P(42, EE), P(44, E_), bas, bas, (int)BT, E_, E_), stream);

    // --- feed forward ---
    ln_rows<<<dim3((unsigned)BT), dim3(256), 0, stream>>>(bas, P(45, E_), P(46, E_), y1, E_);
    run_gemm<false, true, true, false>(wgemm(y1, P(47, (long)E_ * FF_), P(48, FF_), nullptr, hb, (int)BT, FF_, E_), stream);
    run_gemm<false, true, false, true>(wgemm(hb, P(49, (long)FF_ * E_), P(50, E_), bas, bas, (int)BT, E_, FF_), stream);

    // =================== alignment block ===================
    ln_rows<<<dim3((unsigned)BS), dim3(256), 0, stream>>>(sig, P(55, E_), P(56, E_), y1, E_);
    run_gemm<false, false, false, false>(wgemm(y1, P(59, (long)E_ * A_), nullptr, nullptr, sbuf, (int)BS, A_, E_), stream);
    ln_rows<<<dim3((unsigned)BT), dim3(256), 0, stream>>>(bas, P(57, E_), P(58, E_), y1, E_);
    run_gemm<false, false, false, false>(wgemm(y1, P(60, (long)E_ * A_), nullptr, nullptr, bMb, (int)BT, A_, E_), stream);
    aln_gemm<<<dim3((unsigned)(BTS / 64), 2, 1), dim3(256), 0, stream>>>(
        bMb, sbuf, maskp, P(61, (long)A_ * A_), P(62, A_), alnCur, alnNext);

    alnCur = alnNext;
  }
}
